// ResonanceLayer_13761075216777
// MI455X (gfx1250) — matrix operand order repaired and checked by reading, not compiled, not run on hardware
//
#include <hip/hip_runtime.h>
#include <math.h>

// ---------------- problem constants ----------------
#define BB 2
#define TT 4096
#define DD 1024
#define FF 4096
#define SS 4096
#define MM (BB * TT)   // 8192 tokens

typedef __bf16 bf16;
typedef __attribute__((ext_vector_type(16))) __bf16 v16bf;
typedef __attribute__((ext_vector_type(8)))  __bf16 v8bf;
typedef __attribute__((ext_vector_type(8)))  float  v8f;
typedef __attribute__((ext_vector_type(4)))  unsigned int u32x4;
typedef __attribute__((ext_vector_type(8)))  int i32x8;
typedef __attribute__((ext_vector_type(4)))  int i32x4;

// epilogue modes for the WMMA GEMM
#define EP_STORE    0
#define EP_RESID    1   // out = resid + acc
#define EP_BIAS     2   // out = acc + bias[n]
#define EP_BIAS_SIG 3   // out = sigmoid(acc + bias[n])
#define EP_SILU     4   // out = silu(acc)
#define EP_MUL      5   // out = out * acc   (reads existing out)

#if defined(__gfx1250__) && __has_builtin(__builtin_amdgcn_tensor_load_to_lds)
#define USE_TDM 1
#else
#define USE_TDM 0
#endif

__device__ __forceinline__ float sigmoid_(float v) { return 1.0f / (1.0f + expf(-v)); }

#if USE_TDM
// Issue a TDM DMA of a 2D fp32 tile (tile_d0 x tile_d1 elements, row stride
// `stride0` elements) from global memory into LDS at byte offset `lds_off`.
// D# packing per CDNA5 ISA sec 8.3/8.4:
//   group0: [1:0]count=1 | [63:32]lds_addr | [120:64]global_addr | [127:126]type=2
//   group1: [17:16]data_size=2 (4B) | [79:48]tensor_dim0 | [111:80]tensor_dim1
//           | [127:112]tile_dim0 | [143:128]tile_dim1 | [207:160]tensor_dim0_stride
// Groups 2/3 (and the trailing group) zero for a plain 2D tile.
// Completion tracked with TENSORcnt (in-order per wave).
__device__ __forceinline__ void tdm_load_tile_f32(
    unsigned lds_off, const float* gtile, int tensor_d0, int tensor_d1,
    int tile_d0, int tile_d1, int stride0)
{
    unsigned long long ga = (unsigned long long)(uintptr_t)gtile;
    u32x4 g0 = {};
    g0[0] = 1u;                                            // count=1 valid descriptor
    g0[1] = lds_off;                                       // lds_addr (bytes)
    g0[2] = (unsigned)(ga & 0xFFFFFFFFull);                // global_addr[31:0]
    g0[3] = (unsigned)((ga >> 32) & 0x1FFFFFFull) | (2u << 30);  // addr[56:32] | type=2
    i32x8 g1 = {};
    g1[0] = (2 << 16);                                     // data_size = 4 bytes
    g1[1] = (tensor_d0 & 0xFFFF) << 16;                    // tensor_dim0 lo16
    g1[2] = ((tensor_d0 >> 16) & 0xFFFF) | ((tensor_d1 & 0xFFFF) << 16);
    g1[3] = ((tensor_d1 >> 16) & 0xFFFF) | ((tile_d0 & 0xFFFF) << 16);
    g1[4] = tile_d1 & 0xFFFF;                              // tile_dim1 (tile_dim2=0)
    g1[5] = stride0;                                       // tensor_dim0_stride lo32
    g1[6] = 0;                                             // stride hi16 | dim1_stride lo
    g1[7] = 0;
    i32x4 z4 = {};
    i32x8 z8 = {};
    __builtin_amdgcn_tensor_load_to_lds(g0, g1, z4, z4, z8, 0);
}
#endif

// Build WMMA fragments from fp32 LDS tiles (convert to bf16 at read time;
// v_cvt_pk_bf16_f32 co-executes with the WMMA pipe).
// Layouts used here (contiguous k per lane half, not the ISA table order):
//  A 16-bit 16x32: lanes 0-15 -> m=lane, K {0..15}; lanes 16-31 -> K {16..31}
//  B 32x16: n = lane&15, K half by lane>>4, element e -> K = half*16+e
__device__ __forceinline__ void load_frags_f32(const float (*As)[32], const float (*Bs)[32],
                                               int wr, int wc, int lane,
                                               v16bf& a, v16bf& b0, v16bf& b1)
{
    const int m  = wr + (lane & 15);
    const int ko = (lane >> 4) * 16;
#pragma unroll
    for (int i = 0; i < 8; ++i) {
        a[i]     = (bf16)As[m][ko + i];
        a[8 + i] = (bf16)As[m][ko + 8 + i];
    }
    const int n  = lane & 15;
    const int kg = (lane >> 4) * 16;
#pragma unroll
    for (int e = 0; e < 16; ++e) {
        b0[e] = (bf16)Bs[wc + n][kg + e];
        b1[e] = (bf16)Bs[wc + 16 + n][kg + e];
    }
}

// ---------------- WMMA GEMM: Out[M,N] = epilogue(A[M,K] @ W[N,K]^T) ----------------
// Block tile 64x64, K-step 32, 256 threads = 8 waves in a 4x2 grid.
// Double-buffered TDM pipeline: wave 0 issues the DMA pair for tile k+1 into
// the alternate LDS buffer, then waits TENSORcnt<=2 (in-order completion =>
// tile k has landed), barrier publishes, waves compute tile k. DMA latency is
// hidden behind the WMMA body. bf16 conversion happens on fragment read.
__global__ void __launch_bounds__(256)
gemm_wmma(const float* __restrict__ A, const float* __restrict__ W,
          float* __restrict__ Out, const float* __restrict__ resid,
          const float* __restrict__ bias, int M, int N, int K, int mode)
{
    __shared__ __align__(16) float As32[2][64][32];
    __shared__ __align__(16) float Bs32[2][64][32];
    const int tid  = threadIdx.x;
    const int lane = tid & 31;
    const int wave = tid >> 5;
    const int wr   = (wave >> 1) * 16;   // wave row offset in block tile
    const int wc   = (wave & 1) * 32;    // wave col offset in block tile
    const int bm   = blockIdx.y * 64;
    const int bn   = blockIdx.x * 64;

    v8f c0 = {}; v8f c1 = {};

#if USE_TDM
    if (tid < 32) {   // prologue: DMA pair for k-tile 0
        tdm_load_tile_f32((unsigned)(uintptr_t)&As32[0][0][0],
                          &A[(size_t)bm * K], K, M, 32, 64, K);
        tdm_load_tile_f32((unsigned)(uintptr_t)&Bs32[0][0][0],
                          &W[(size_t)bn * K], K, N, 32, 64, K);
    }
    int idx = 0;
    for (int k0 = 0; k0 < K; k0 += 32, idx ^= 1) {
        if (tid < 32) {
            if (k0 + 32 < K) {   // issue next pair into the other buffer
                tdm_load_tile_f32((unsigned)(uintptr_t)&As32[idx ^ 1][0][0],
                                  &A[(size_t)bm * K + k0 + 32], K, M, 32, 64, K);
                tdm_load_tile_f32((unsigned)(uintptr_t)&Bs32[idx ^ 1][0][0],
                                  &W[(size_t)bn * K + k0 + 32], K, N, 32, 64, K);
                __builtin_amdgcn_s_wait_tensorcnt(2);   // current pair complete
            } else {
                __builtin_amdgcn_s_wait_tensorcnt(0);   // final pair complete
            }
        }
        __syncthreads();
        v16bf a, b0, b1;
        load_frags_f32(As32[idx], Bs32[idx], wr, wc, lane, a, b0, b1);
        c0 = __builtin_amdgcn_wmma_f32_16x16x32_bf16(false, a, false, b0, (short)0, c0, false, false);
        c1 = __builtin_amdgcn_wmma_f32_16x16x32_bf16(false, a, false, b1, (short)0, c1, false, false);
        __syncthreads();   // all reads of buf[idx] done before it is re-filled
    }
#else
    for (int k0 = 0; k0 < K; k0 += 32) {
        {   // vectorized fallback: 2x b128 global load + 2x b128 LDS store / thread
            const int sr = tid >> 2, sc = (tid & 3) << 3;
            *(float4*)&As32[0][sr][sc]     = *(const float4*)&A[(size_t)(bm + sr) * K + k0 + sc];
            *(float4*)&As32[0][sr][sc + 4] = *(const float4*)&A[(size_t)(bm + sr) * K + k0 + sc + 4];
            *(float4*)&Bs32[0][sr][sc]     = *(const float4*)&W[(size_t)(bn + sr) * K + k0 + sc];
            *(float4*)&Bs32[0][sr][sc + 4] = *(const float4*)&W[(size_t)(bn + sr) * K + k0 + sc + 4];
        }
        __syncthreads();
        v16bf a, b0, b1;
        load_frags_f32(As32[0], Bs32[0], wr, wc, lane, a, b0, b1);
        c0 = __builtin_amdgcn_wmma_f32_16x16x32_bf16(false, a, false, b0, (short)0, c0, false, false);
        c1 = __builtin_amdgcn_wmma_f32_16x16x32_bf16(false, a, false, b1, (short)0, c1, false, false);
        __syncthreads();
    }
#endif

    // C layout: VGPR r -> M = wr + 8*(lane>>4) + r, N = (lane&15)
    const int rowoff = bm + wr + ((lane >> 4) << 3);
    const int col0   = bn + wc + (lane & 15);
#pragma unroll
    for (int r = 0; r < 8; ++r) {
        const int row = rowoff + r;
        const int cA = col0, cB = col0 + 16;
        const size_t oA = (size_t)row * N + cA;
        const size_t oB = (size_t)row * N + cB;
        float vA = c0[r], vB = c1[r];
        switch (mode) {
            case EP_STORE:    Out[oA] = vA;                       Out[oB] = vB;                       break;
            case EP_RESID:    Out[oA] = resid[oA] + vA;           Out[oB] = resid[oB] + vB;           break;
            case EP_BIAS:     Out[oA] = vA + bias[cA];            Out[oB] = vB + bias[cB];            break;
            case EP_BIAS_SIG: Out[oA] = sigmoid_(vA + bias[cA]);  Out[oB] = sigmoid_(vB + bias[cB]);  break;
            case EP_SILU:     Out[oA] = vA * sigmoid_(vA);        Out[oB] = vB * sigmoid_(vB);        break;
            case EP_MUL:      Out[oA] *= vA;                      Out[oB] *= vB;                      break;
        }
    }
}

// ---------------- fused scores GEMM + top-2 routing ----------------
// Each block owns 64 rows, streams all S slot columns in 64-wide tiles through
// WMMA, dumps the scaled 64x64 score tile to LDS, and row-owner threads keep a
// running top-2 (value,index) in registers. Avoids materializing [M,S] scores.
// Staging here is manual (vectorized b128 path) as a non-TDM variant.
__global__ void __launch_bounds__(256)
scores_topk(const float* __restrict__ Q, const float* __restrict__ SK,
            float* __restrict__ topv, int* __restrict__ topi,
            int M, int S, int K, float scale)
{
    __shared__ __align__(16) float As32[64][32];
    __shared__ __align__(16) float Bs32[64][32];
    __shared__ float Sc[64][65];
    const int tid  = threadIdx.x;
    const int lane = tid & 31;
    const int wave = tid >> 5;
    const int wr   = (wave >> 1) * 16;
    const int wc   = (wave & 1) * 32;
    const int bm   = blockIdx.x * 64;
    const int sr   = tid >> 2, sc = (tid & 3) << 3;

    float b1v = -3.4e38f, b2v = -3.4e38f;
    int   b1i = 0,        b2i = 0;

    for (int n0 = 0; n0 < S; n0 += 64) {
        v8f c0 = {}; v8f c1 = {};
        for (int k0 = 0; k0 < K; k0 += 32) {
            *(float4*)&As32[sr][sc]     = *(const float4*)&Q [(size_t)(bm + sr) * K + k0 + sc];
            *(float4*)&As32[sr][sc + 4] = *(const float4*)&Q [(size_t)(bm + sr) * K + k0 + sc + 4];
            *(float4*)&Bs32[sr][sc]     = *(const float4*)&SK[(size_t)(n0 + sr) * K + k0 + sc];
            *(float4*)&Bs32[sr][sc + 4] = *(const float4*)&SK[(size_t)(n0 + sr) * K + k0 + sc + 4];
            __syncthreads();
            v16bf a, b0, b1;
            load_frags_f32(As32, Bs32, wr, wc, lane, a, b0, b1);
            c0 = __builtin_amdgcn_wmma_f32_16x16x32_bf16(false, a, false, b0, (short)0, c0, false, false);
            c1 = __builtin_amdgcn_wmma_f32_16x16x32_bf16(false, a, false, b1, (short)0, c1, false, false);
            __syncthreads();
        }
        const int rl = wr + ((lane >> 4) << 3);
        const int cl = wc + (lane & 15);
#pragma unroll
        for (int r = 0; r < 8; ++r) {
            Sc[rl + r][cl]      = c0[r] * scale;
            Sc[rl + r][cl + 16] = c1[r] * scale;
        }
        __syncthreads();
        if (tid < 64) {  // row-owner scan; strict > keeps earliest index on ties
            for (int j = 0; j < 64; ++j) {
                const float v = Sc[tid][j];
                const int   c = n0 + j;
                if (v > b1v)      { b2v = b1v; b2i = b1i; b1v = v; b1i = c; }
                else if (v > b2v) { b2v = v; b2i = c; }
            }
        }
        __syncthreads();
    }
    if (tid < 64) {
        const int m = bm + tid;
        topv[2 * m] = b1v; topv[2 * m + 1] = b2v;
        topi[2 * m] = b1i; topi[2 * m + 1] = b2i;
    }
}

// ---------------- RMSNorm (vectorized: D=1024 -> one float4 per thread) ----------------
__global__ void __launch_bounds__(256)
rmsnorm_k(const float* __restrict__ X, const float* __restrict__ w,
          float* __restrict__ O, int D)
{
    __shared__ float red[256];
    const int row = blockIdx.x, tid = threadIdx.x;
    const float4* x4 = (const float4*)(X + (size_t)row * D);
    const float4* w4 = (const float4*)w;
    float4*       o4 = (float4*)(O + (size_t)row * D);
    const int n4 = D >> 2;
    float ss = 0.f;
    for (int i = tid; i < n4; i += 256) {
        const float4 v = x4[i];
        ss += v.x * v.x + v.y * v.y + v.z * v.z + v.w * v.w;
    }
    red[tid] = ss; __syncthreads();
    for (int s = 128; s > 0; s >>= 1) { if (tid < s) red[tid] += red[tid + s]; __syncthreads(); }
    const float inv = 1.0f / sqrtf(red[0] / (float)D + 1e-6f);
    for (int i = tid; i < n4; i += 256) {
        const float4 v = x4[i], g = w4[i];
        float4 r;
        r.x = v.x * g.x * inv; r.y = v.y * g.y * inv;
        r.z = v.z * g.z * inv; r.w = v.w * g.w * inv;
        o4[i] = r;
    }
}

// ---------------- multi-scale causal depthwise conv (k=3,7,15 summed) ----------------
__global__ void __launch_bounds__(256)
conv_k(const float* __restrict__ xn, const float* __restrict__ w3,
       const float* __restrict__ w7, const float* __restrict__ w15,
       float* __restrict__ out, int T, int D)
{
    const int m = blockIdx.x;          // token index in [0, B*T)
    const int t = m % T;               // position in its sequence (causal bound)
    const float* base = xn + (size_t)m * D;
    float* o = out + (size_t)m * D;
    const int maxoff = t < 14 ? t : 14;
    for (int d = threadIdx.x; d < D; d += 256) {
        float acc = 0.f;
        for (int off = 0; off <= maxoff; ++off) {
            float wv = w15[d * 15 + (14 - off)];
            if (off < 7) wv += w7[d * 7 + (6 - off)];
            if (off < 3) wv += w3[d * 3 + (2 - off)];
            acc += base[(long)d - (long)off * D] * wv;
        }
        o[d] = acc;
    }
}

// ---------------- MinGRU linear recurrence (serial over T, parallel over B*D) ----------------
__global__ void __launch_bounds__(256)
mingru_scan(const float* __restrict__ z, const float* __restrict__ ht,
            const float* __restrict__ h0, float* __restrict__ x_io,
            float* __restrict__ h_last, int T, int D)
{
    const int ch = blockIdx.x * blockDim.x + threadIdx.x;   // [0, B*D)
    const int b = ch / D, d = ch - b * D;
    float h = h0[ch];
    const size_t base = (size_t)b * T * D + d;
    for (int t = 0; t < T; ++t) {
        const size_t idx = base + (size_t)t * D;
        const float zt = z[idx], hv = ht[idx];
        h = (1.0f - zt) * h + zt * hv;
        x_io[idx] += h;
    }
    h_last[ch] = h;
}

// ---------------- top-2 softmax + slot-value gather ----------------
__global__ void __launch_bounds__(256)
retrieve_k(const float* __restrict__ topv, const int* __restrict__ topi,
           const float* __restrict__ V, float* __restrict__ out, int D)
{
    const int m = blockIdx.x;
    const float v0 = topv[2 * m], v1 = topv[2 * m + 1];
    const float mx = fmaxf(v0, v1);
    const float e0 = expf(v0 - mx), e1 = expf(v1 - mx);
    const float inv = 1.0f / (e0 + e1);
    const float a0 = e0 * inv, a1 = e1 * inv;
    const float* V0 = V + (size_t)topi[2 * m] * D;
    const float* V1 = V + (size_t)topi[2 * m + 1] * D;
    float* o = out + (size_t)m * D;
    for (int i = threadIdx.x; i < D; i += 256)
        o[i] = a0 * V0[i] + a1 * V1[i];
}

// ---------------- host orchestration ----------------
static inline void launch_gemm(const float* A, const float* W, float* Out,
                               const float* resid, const float* bias,
                               int M, int N, int K, int mode, hipStream_t s)
{
    dim3 grid(N / 64, M / 64);
    gemm_wmma<<<grid, 256, 0, s>>>(A, W, Out, resid, bias, M, N, K, mode);
}

extern "C" void kernel_launch(void* const* d_in, const int* in_sizes, int n_in,
                              void* d_out, int out_size, void* d_ws, size_t ws_size,
                              hipStream_t stream)
{
    const float* x        = (const float*)d_in[0];
    const float* h_prev   = (const float*)d_in[1];
    const float* norm1_w  = (const float*)d_in[2];
    const float* conv3_w  = (const float*)d_in[3];
    const float* conv7_w  = (const float*)d_in[4];
    const float* conv15_w = (const float*)d_in[5];
    const float* mix_w    = (const float*)d_in[6];
    const float* norm2_w  = (const float*)d_in[7];
    const float* Wz_w     = (const float*)d_in[8];
    const float* Wz_b     = (const float*)d_in[9];
    const float* Wh_w     = (const float*)d_in[10];
    const float* Wh_b     = (const float*)d_in[11];
    const float* norm3_w  = (const float*)d_in[12];
    const float* slot_k   = (const float*)d_in[13];
    const float* slot_v   = (const float*)d_in[14];
    const float* projq_w  = (const float*)d_in[15];
    const float* projout_w= (const float*)d_in[16];
    const float* norm4_w  = (const float*)d_in[17];
    const float* gate_w   = (const float*)d_in[18];
    const float* up_w     = (const float*)d_in[19];
    const float* down_w   = (const float*)d_in[20];

    float* out_x = (float*)d_out;                  // [M, D]
    float* out_h = out_x + (size_t)MM * DD;        // [B, D]

    char* ws = (char*)d_ws;
    const size_t MD = (size_t)MM * DD * sizeof(float);
    float* xn   = (float*)(ws);            // rmsnorm output
    float* tmp1 = (float*)(ws + 1 * MD);   // conv / z / retrieved
    float* tmp2 = (float*)(ws + 2 * MD);   // h_tilde / q
    float* xcur = (float*)(ws + 3 * MD);   // running residual x
    float* ffb  = (float*)(ws + 4 * MD);   // FFN chunk buffer [2048, F]
    float* topv = (float*)(ws + 5 * MD);
    int*   topi = (int*)  (ws + 5 * MD + (size_t)MM * 2 * sizeof(float));

    // ---- stage 1: multi-scale conv block + mix projection (residual) ----
    rmsnorm_k<<<MM, 256, 0, stream>>>(x, norm1_w, xn, DD);
    conv_k<<<MM, 256, 0, stream>>>(xn, conv3_w, conv7_w, conv15_w, tmp1, TT, DD);
    launch_gemm(tmp1, mix_w, xcur, x, nullptr, MM, DD, DD, EP_RESID, stream);

    // ---- stage 2: MinGRU ----
    rmsnorm_k<<<MM, 256, 0, stream>>>(xcur, norm2_w, xn, DD);
    launch_gemm(xn, Wz_w, tmp1, nullptr, Wz_b, MM, DD, DD, EP_BIAS_SIG, stream);  // z
    launch_gemm(xn, Wh_w, tmp2, nullptr, Wh_b, MM, DD, DD, EP_BIAS, stream);      // h_tilde
    mingru_scan<<<(BB * DD) / 256, 256, 0, stream>>>(tmp1, tmp2, h_prev, xcur, out_h, TT, DD);

    // ---- stage 3: slot memory with top-2 routing ----
    rmsnorm_k<<<MM, 256, 0, stream>>>(xcur, norm3_w, xn, DD);
    launch_gemm(xn, projq_w, tmp2, nullptr, nullptr, MM, DD, DD, EP_STORE, stream);   // q
    scores_topk<<<MM / 64, 256, 0, stream>>>(tmp2, slot_k, topv, topi, MM, SS, DD, 0.03125f);
    retrieve_k<<<MM, 256, 0, stream>>>(topv, topi, slot_v, tmp1, DD);
    launch_gemm(tmp1, projout_w, xcur, xcur, nullptr, MM, DD, DD, EP_RESID, stream);

    // ---- stage 4: SwiGLU FFN, chunked over rows to bound workspace ----
    rmsnorm_k<<<MM, 256, 0, stream>>>(xcur, norm4_w, xn, DD);
    for (int c = 0; c < 4; ++c) {
        const size_t roff = (size_t)c * 2048;
        launch_gemm(xn + roff * DD, gate_w, ffb, nullptr, nullptr, 2048, FF, DD, EP_SILU, stream);
        launch_gemm(xn + roff * DD, up_w,   ffb, nullptr, nullptr, 2048, FF, DD, EP_MUL,  stream);
        launch_gemm(ffb, down_w, out_x + roff * DD, xcur + roff * DD, nullptr,
                    2048, DD, FF, EP_RESID, stream);
    }
}
